// DGCNN_semseg_s3dis_54185307406637
// MI455X (gfx1250) — matrix operand order repaired and checked by reading, not compiled, not run on hardware
//
#include <hip/hip_runtime.h>
#include <hip/hip_bf16.h>

#define EPS    1e-5f
#define SLOPE  0.2f
#define BATCH  8
#define CIN    9
#define NPTS   4096
#define KMAX   20
#define NEG_BIG (-3.402823466e38f)

typedef __attribute__((ext_vector_type(16))) _Float16 v16h;
typedef __attribute__((ext_vector_type(8)))  _Float16 v8h;
typedef __attribute__((ext_vector_type(8)))  float    v8f;

__device__ __forceinline__ float lrelu(float x) { return x >= 0.f ? x : SLOPE * x; }

// ---------------- WMMA tile helpers (16x16x32 f16, wave32) ----------------
// A 16x32 f16: lane L -> M=L&15; lanes<16 hold K = kk+0..15,
// lanes>=16 hold K = kk+16..31; packed pairs, low half = even K.
__device__ __forceinline__ v16h load_a_h(const _Float16* __restrict__ Wh, int Cin,
                                         int orow, int kk, int lane) {
  int m  = lane & 15;
  int kb = (lane < 16) ? 0 : 16;
  const _Float16* row = Wh + (size_t)(orow + m) * Cin;
  union { v16h v; v8h h[2]; } u;
  u.h[0] = *(const v8h*)(row + kk + kb);
  u.h[1] = *(const v8h*)(row + kk + 8 + kb);
  return u.v;
}

// B 32x16 f16: lane -> col N=lane&15; lanes<16 rows K=kk+0..15, lanes>=16 rows
// K=kk+16..31 (packed pairs). One 32-byte contiguous load per lane.
__device__ __forceinline__ v16h load_b_h(const _Float16* __restrict__ Bt, int ldb,
                                         int col0, int kofs, int kk, int lane) {
  int n  = lane & 15;
  int kb = (lane < 16) ? 0 : 16;
  return *(const v16h*)(Bt + (size_t)(col0 + n) * ldb + kofs + kk + kb);
}

// ---------------- misc ----------------
__global__ void zero_kernel(float* __restrict__ p, int n) {
  int i = blockIdx.x * 256 + threadIdx.x;
  if (i < n) p[i] = 0.f;
}

__global__ void wconv_kernel(const float* __restrict__ W, _Float16* __restrict__ Wh, int n) {
  int i = blockIdx.x * 256 + threadIdx.x;
  if (i < n) Wh[i] = (_Float16)W[i];
}

// w9 (13x256) zero-padded to 16x256 f16
__global__ void w9conv_kernel(const float* __restrict__ W, _Float16* __restrict__ Wh) {
  int i = blockIdx.x * 256 + threadIdx.x;
  if (i < 16 * 256) {
    int o = i >> 8, c = i & 255;
    Wh[i] = (o < 13) ? (_Float16)W[o * 256 + c] : (_Float16)0.f;
  }
}

// ---------------- kNN: one wave per (b,n) row, sorted top-20 ----------------
__global__ __launch_bounds__(32) void knn_kernel(const float* __restrict__ x,
                                                 int* __restrict__ idx20) {
  int p = blockIdx.x;                 // 0 .. BATCH*NPTS-1
  int b = p >> 12, n = p & 4095;
  int lane = threadIdx.x;
  const float* xb = x + (size_t)b * CIN * NPTS;
  float c9[CIN]; float xxn = 0.f;
#pragma unroll
  for (int c = 0; c < CIN; ++c) { c9[c] = xb[c*NPTS + n]; xxn += c9[c]*c9[c]; }

  float tv[KMAX]; int ti[KMAX];
#pragma unroll
  for (int i = 0; i < KMAX; ++i) { tv[i] = NEG_BIG; ti[i] = 0; }

  for (int m = lane; m < NPTS; m += 32) {
    float dot = 0.f, xxm = 0.f;
#pragma unroll
    for (int c = 0; c < CIN; ++c) { float v = xb[c*NPTS + m]; dot += c9[c]*v; xxm += v*v; }
    float d = 2.f*dot - xxm - xxn;    // neg squared distance
    if (d > tv[KMAX-1]) {
      int q = KMAX - 1;
      while (q > 0 && tv[q-1] < d) { tv[q] = tv[q-1]; ti[q] = ti[q-1]; --q; }
      tv[q] = d; ti[q] = m;
    }
  }
  __shared__ float lv[32][KMAX];
  __shared__ int   li[32][KMAX];
#pragma unroll
  for (int i = 0; i < KMAX; ++i) { lv[lane][i] = tv[i]; li[lane][i] = ti[i]; }
  __syncthreads();
  int ptr = 0;
  for (int r = 0; r < KMAX; ++r) {
    float v = (ptr < KMAX) ? lv[lane][ptr] : NEG_BIG;
    int who = lane;
#pragma unroll
    for (int off = 16; off > 0; off >>= 1) {
      float ov = __shfl_xor(v, off, 32);
      int   ow = __shfl_xor(who, off, 32);
      if (ov > v || (ov == v && ow < who)) { v = ov; who = ow; }
    }
    int wptr = __shfl(ptr, who, 32);
    int widx = li[who][wptr];
    if (lane == 0) idx20[(size_t)p * KMAX + r] = widx;
    if (lane == who) ptr++;
  }
}

// ---------------- edge feature f0 (18) ----------------
__device__ __forceinline__ void make_f0(const float* __restrict__ xb, int n, int nb,
                                        float* f0) {
#pragma unroll
  for (int c = 0; c < CIN; ++c) {
    float ce = xb[c*NPTS + n];
    float nv = xb[c*NPTS + nb];
    f0[c]       = nv - ce;
    f0[c + CIN] = ce;
  }
}

// ---------------- moment pass, stage 0: E[f0], E[f0 f0^T] ----------------
__global__ __launch_bounds__(256) void edge_mom0_kernel(const float* __restrict__ x,
    const int* __restrict__ idx20, int k, float* __restrict__ mom /*[324 M][18 mu]*/) {
  __shared__ float F[256][18];
  int tid = threadIdx.x;
  long S = (long)BATCH * NPTS * k;
  long s = (long)blockIdx.x * 256 + tid;
  float f0[18];
  if (s < S) {
    int j = (int)(s % k); int p = (int)(s / k);
    int b = p >> 12, n = p & 4095;
    make_f0(x + (size_t)b*CIN*NPTS, n, idx20[(size_t)p*KMAX + j], f0);
  } else {
#pragma unroll
    for (int c = 0; c < 18; ++c) f0[c] = 0.f;
  }
#pragma unroll
  for (int c = 0; c < 18; ++c) F[tid][c] = f0[c];
  __syncthreads();
  for (int e = tid; e < 18*18 + 18; e += 256) {
    float acc = 0.f;
    if (e < 324) { int i = e/18, j = e%18; for (int ss = 0; ss < 256; ++ss) acc += F[ss][i]*F[ss][j]; }
    else         { int c = e - 324;        for (int ss = 0; ss < 256; ++ss) acc += F[ss][c]; }
    atomicAdd(&mom[e], acc);
  }
}

// ---------------- moment pass, stage 1: moments of z1 (32) ----------------
__global__ __launch_bounds__(128) void edge_mom1_kernel(const float* __restrict__ x,
    const int* __restrict__ idx20, int k,
    const float* __restrict__ W1, const float* __restrict__ st1,
    float* __restrict__ mom /*[1024][32]*/) {
  __shared__ float Z[128][32];
  __shared__ float lW1[576], l1[64];
  int tid = threadIdx.x;
  for (int e = tid; e < 576; e += 128) lW1[e] = W1[e];
  if (tid < 64) l1[tid] = st1[tid];
  __syncthreads();
  long S = (long)BATCH * NPTS * k;
  long s = (long)blockIdx.x * 128 + tid;
  float z1[32];
  if (s < S) {
    int j = (int)(s % k); int p = (int)(s / k);
    int b = p >> 12, n = p & 4095;
    float f0[18];
    make_f0(x + (size_t)b*CIN*NPTS, n, idx20[(size_t)p*KMAX + j], f0);
    for (int o = 0; o < 32; ++o) {
      float a = 0.f;
      for (int c = 0; c < 18; ++c) a += lW1[o*18 + c]*f0[c];
      z1[o] = lrelu((a - l1[2*o]) * l1[2*o+1]);
    }
  } else { for (int o = 0; o < 32; ++o) z1[o] = 0.f; }
  for (int o = 0; o < 32; ++o) Z[tid][o] = z1[o];
  __syncthreads();
  for (int e = tid; e < 32*32 + 32; e += 128) {
    float acc = 0.f;
    if (e < 1024) { int i = e/32, j = e%32; for (int ss = 0; ss < 128; ++ss) acc += Z[ss][i]*Z[ss][j]; }
    else          { int c = e - 1024;       for (int ss = 0; ss < 128; ++ss) acc += Z[ss][c]; }
    atomicAdd(&mom[e], acc);
  }
}

// ---------------- moment pass, stage 2: moments of z2 (64) ----------------
__global__ __launch_bounds__(128) void edge_mom2_kernel(const float* __restrict__ x,
    const int* __restrict__ idx20, int k,
    const float* __restrict__ W1, const float* __restrict__ W11,
    const float* __restrict__ st1, const float* __restrict__ st2,
    float* __restrict__ mom /*[4096][64]*/) {
  __shared__ float Z[128][64];
  __shared__ float lW1[576], lW11[2048], l1[64], l2[128];
  int tid = threadIdx.x;
  for (int e = tid; e < 576;  e += 128) lW1[e]  = W1[e];
  for (int e = tid; e < 2048; e += 128) lW11[e] = W11[e];
  if (tid < 64)  l1[tid] = st1[tid];
  if (tid < 128) l2[tid] = st2[tid];
  __syncthreads();
  long S = (long)BATCH * NPTS * k;
  long s = (long)blockIdx.x * 128 + tid;
  float z2[64];
  if (s < S) {
    int j = (int)(s % k); int p = (int)(s / k);
    int b = p >> 12, n = p & 4095;
    float f0[18];
    make_f0(x + (size_t)b*CIN*NPTS, n, idx20[(size_t)p*KMAX + j], f0);
    float z1[32];
    for (int o = 0; o < 32; ++o) {
      float a = 0.f;
      for (int c = 0; c < 18; ++c) a += lW1[o*18 + c]*f0[c];
      z1[o] = lrelu((a - l1[2*o]) * l1[2*o+1]);
    }
    for (int o = 0; o < 64; ++o) {
      float a = 0.f;
      for (int c = 0; c < 32; ++c) a += lW11[o*32 + c]*z1[c];
      z2[o] = lrelu((a - l2[2*o]) * l2[2*o+1]);
    }
  } else { for (int o = 0; o < 64; ++o) z2[o] = 0.f; }
  for (int o = 0; o < 64; ++o) Z[tid][o] = z2[o];
  __syncthreads();
  for (int e = tid; e < 64*64 + 64; e += 128) {
    float acc = 0.f;
    if (e < 4096) { int i = e/64, j = e%64; for (int ss = 0; ss < 128; ++ss) acc += Z[ss][i]*Z[ss][j]; }
    else          { int c = e - 4096;       for (int ss = 0; ss < 128; ++ss) acc += Z[ss][c]; }
    atomicAdd(&mom[e], acc);
  }
}

// ---- BN stats of y=W*f from input moments: mean=W*mu/S, E[y^2]=W M W^T/S ----
__global__ __launch_bounds__(64) void linstats_kernel(const float* __restrict__ W,
    int Cin, int Cout, const float* __restrict__ mom, float Sinv,
    float* __restrict__ st /* interleaved mean,rstd */) {
  int o = blockIdx.x * 64 + threadIdx.x;
  if (o >= Cout) return;
  const float* Wo = W + (size_t)o * Cin;
  const float* M  = mom;
  const float* mu = mom + (size_t)Cin * Cin;
  float mean = 0.f;
  for (int c = 0; c < Cin; ++c) mean += Wo[c] * mu[c];
  mean *= Sinv;
  float q = 0.f;
  for (int c1 = 0; c1 < Cin; ++c1) {
    float w1v = Wo[c1];
    const float* Mr = M + (size_t)c1 * Cin;
    for (int c2 = 0; c2 < Cin; ++c2) q += w1v * Wo[c2] * Mr[c2];
  }
  float var = q * Sinv - mean * mean;
  st[2*o]   = mean;
  st[2*o+1] = rsqrtf(var + EPS);
}

// ---------------- fused edge branch: full 3-layer chain + max over k ----------------
__global__ __launch_bounds__(128) void edge_final_kernel(const float* __restrict__ x,
    const int* __restrict__ idx20, int k,
    const float* __restrict__ W1, const float* __restrict__ W11, const float* __restrict__ W2,
    const float* __restrict__ st1, const float* __restrict__ st2, const float* __restrict__ st3,
    float* __restrict__ s192, int coff) {
  __shared__ float lW1[576], lW11[2048], lW2[4096], l1[64], l2[128], l3[128];
  int tid = threadIdx.x;
  for (int e = tid; e < 576;  e += 128) lW1[e]  = W1[e];
  for (int e = tid; e < 2048; e += 128) lW11[e] = W11[e];
  for (int e = tid; e < 4096; e += 128) lW2[e]  = W2[e];
  if (tid < 64)  l1[tid] = st1[tid];
  if (tid < 128) { l2[tid] = st2[tid]; l3[tid] = st3[tid]; }
  __syncthreads();
  int p = blockIdx.x * 128 + tid;          // 0..32767
  int b = p >> 12, n = p & 4095;
  const float* xb = x + (size_t)b * CIN * NPTS;
  float mx[64];
  for (int o = 0; o < 64; ++o) mx[o] = NEG_BIG;
  for (int j = 0; j < k; ++j) {
    float f0[18];
    make_f0(xb, n, idx20[(size_t)p*KMAX + j], f0);
    float z1[32];
    for (int o = 0; o < 32; ++o) {
      float a = 0.f;
      for (int c = 0; c < 18; ++c) a += lW1[o*18 + c]*f0[c];
      z1[o] = lrelu((a - l1[2*o]) * l1[2*o+1]);
    }
    float z2[64];
    for (int o = 0; o < 64; ++o) {
      float a = 0.f;
      for (int c = 0; c < 32; ++c) a += lW11[o*32 + c]*z1[c];
      z2[o] = lrelu((a - l2[2*o]) * l2[2*o+1]);
    }
    for (int o = 0; o < 64; ++o) {
      float a = 0.f;
      for (int c = 0; c < 64; ++c) a += lW2[o*64 + c]*z2[c];
      float z3 = lrelu((a - l3[2*o]) * l3[2*o+1]);
      mx[o] = fmaxf(mx[o], z3);
    }
  }
  for (int o = 0; o < 64; ++o)
    s192[((size_t)b*192 + coff + o)*NPTS + n] = mx[o];
}

// ---------------- 192x192 second moment of concat(x1,x2,x3) ----------------
__global__ __launch_bounds__(256) void mom192_kernel(const float* __restrict__ s192,
                                                     float* __restrict__ mom) {
  int e = blockIdx.x * 256 + threadIdx.x;
  const int E = 192*192 + 192;
  if (e >= E) return;
  int s0 = blockIdx.y * 2048;
  float acc = 0.f;
  if (e < 192*192) {
    int i = e / 192, j = e % 192;
    for (int s = s0; s < s0 + 2048; ++s) {
      int b = s >> 12, n = s & 4095;
      acc += s192[((size_t)b*192 + i)*NPTS + n] * s192[((size_t)b*192 + j)*NPTS + n];
    }
  } else {
    int c = e - 192*192;
    for (int s = s0; s < s0 + 2048; ++s) {
      int b = s >> 12, n = s & 4095;
      acc += s192[((size_t)b*192 + c)*NPTS + n];
    }
  }
  atomicAdd(&mom[e], acc);
}

// -------- s192 -> tail of hT (f16, K-innermost): hT[b][n][1024+c] --------
__global__ void s192_to_hT_kernel(const float* __restrict__ s192, _Float16* __restrict__ hT) {
  int e = blockIdx.x * 256 + threadIdx.x;          // over BATCH*NPTS*192
  if (e >= BATCH * NPTS * 192) return;
  int c = e % 192; int p = e / 192;
  int b = p >> 12, n = p & 4095;
  hT[(size_t)p * 1216 + 1024 + c] = (_Float16)s192[((size_t)b*192 + c)*NPTS + n];
}

// -------- g broadcast -> head of hT: hT[b][n][c] = g[b][c] --------
__global__ void g_to_hT_kernel(const float* __restrict__ g, _Float16* __restrict__ hT) {
  int e = blockIdx.x * 256 + threadIdx.x;          // over BATCH*NPTS*1024
  if (e >= BATCH * NPTS * 1024) return;
  int c = e & 1023; int p = e >> 10;
  int b = p >> 12;
  hT[(size_t)p * 1216 + c] = (_Float16)g[(size_t)b*1024 + c];
}

// -------- normalize + lrelu + transpose to f16 K-innermost --------
__global__ void normT_kernel(const float* __restrict__ y, const float* __restrict__ st,
                             int C, _Float16* __restrict__ zT) {
  int e = blockIdx.x * 256 + threadIdx.x;          // over BATCH*NPTS*C
  if (e >= BATCH * NPTS * C) return;
  int c = e % C; int p = e / C;
  int b = p >> 12, n = p & 4095;
  float raw = y[((size_t)b*C + c)*NPTS + n];
  zT[(size_t)p * C + c] = (_Float16)lrelu((raw - st[2*c]) * st[2*c+1]);
}

// ---------------- w6 WMMA GEMM + BN + lrelu + max over n -> g[b][1024] ----------------
// Each wave: contiguous strip of 64 column tiles, processed 4 at a time with
// 4 independent accumulators (A fully hoisted).
__global__ __launch_bounds__(128) void w6_max_kernel(const _Float16* __restrict__ hT,
    const _Float16* __restrict__ w6h, const float* __restrict__ st6, float* __restrict__ g) {
  int b = blockIdx.x;
  int orow = blockIdx.y * 16;
  int wave = threadIdx.x >> 5, lane = threadIdx.x & 31;
  v16h a[6];
#pragma unroll
  for (int t = 0; t < 6; ++t) a[t] = load_a_h(w6h, 192, orow, t*32, lane);
  float rmax[8];
#pragma unroll
  for (int i = 0; i < 8; ++i) rmax[i] = NEG_BIG;
  const _Float16* Bt = hT + (size_t)b * NPTS * 1216;
  int ntbase = wave * 64;
  for (int nt = ntbase; nt < ntbase + 64; nt += 4) {
    int col0 = nt * 16;
    v8f acc0 = {}, acc1 = {}, acc2 = {}, acc3 = {};
#pragma unroll
    for (int t = 0; t < 6; ++t) {
      int kk = t * 32;
      v16h b0 = load_b_h(Bt, 1216, col0,      1024, kk, lane);
      v16h b1 = load_b_h(Bt, 1216, col0 + 16, 1024, kk, lane);
      v16h b2 = load_b_h(Bt, 1216, col0 + 32, 1024, kk, lane);
      v16h b3 = load_b_h(Bt, 1216, col0 + 48, 1024, kk, lane);
      acc0 = __builtin_amdgcn_wmma_f32_16x16x32_f16(false, a[t], false, b0, (short)0, acc0, false, false);
      acc1 = __builtin_amdgcn_wmma_f32_16x16x32_f16(false, a[t], false, b1, (short)0, acc1, false, false);
      acc2 = __builtin_amdgcn_wmma_f32_16x16x32_f16(false, a[t], false, b2, (short)0, acc2, false, false);
      acc3 = __builtin_amdgcn_wmma_f32_16x16x32_f16(false, a[t], false, b3, (short)0, acc3, false, false);
    }
#pragma unroll
    for (int i = 0; i < 8; ++i) {
      int m = (lane < 16) ? i : (8 + i);
      int o = orow + m;
      float mn = st6[2*o], rs = st6[2*o+1];
      float v01 = fmaxf(lrelu((acc0[i] - mn) * rs), lrelu((acc1[i] - mn) * rs));
      float v23 = fmaxf(lrelu((acc2[i] - mn) * rs), lrelu((acc3[i] - mn) * rs));
      rmax[i] = fmaxf(rmax[i], fmaxf(v01, v23));
    }
  }
#pragma unroll
  for (int off = 8; off > 0; off >>= 1) {
#pragma unroll
    for (int i = 0; i < 8; ++i) rmax[i] = fmaxf(rmax[i], __shfl_xor(rmax[i], off, 32));
  }
  __shared__ float wm[4][16];
  if (lane == 0)  { for (int i = 0; i < 8; ++i) wm[wave][i]     = rmax[i]; }
  if (lane == 16) { for (int i = 0; i < 8; ++i) wm[wave][8 + i] = rmax[i]; }
  __syncthreads();
  if (threadIdx.x < 16) {
    float v = fmaxf(fmaxf(wm[0][threadIdx.x], wm[1][threadIdx.x]),
                    fmaxf(wm[2][threadIdx.x], wm[3][threadIdx.x]));
    g[(size_t)b*1024 + orow + threadIdx.x] = v;
  }
}

// ---------------- w7 WMMA GEMM: hT (1216) -> y7 raw f32 (512) ----------------
// 4 column tiles per wave: A amortized 4x, 4 independent WMMA chains.
__global__ __launch_bounds__(128) void w7_gemm_kernel(const _Float16* __restrict__ hT,
    const _Float16* __restrict__ w7h, float* __restrict__ y7) {
  int b = blockIdx.x;
  int orow = blockIdx.y * 16;
  int wave = threadIdx.x >> 5, lane = threadIdx.x & 31;
  int col0 = (blockIdx.z * 4 + wave) * 64;       // 4 tiles of 16 cols
  const _Float16* Bt = hT + (size_t)b * NPTS * 1216;
  const _Float16* bRow = Bt + (size_t)(col0 + (lane & 15)) * 1216;
  v8f acc0 = {}, acc1 = {}, acc2 = {}, acc3 = {};
  for (int t = 0; t < 38; ++t) {
    int kk = t * 32;
    __builtin_prefetch((const void*)(bRow + kk + 64), 0, 0);
    v16h a  = load_a_h(w7h, 1216, orow, kk, lane);
    v16h b0 = load_b_h(Bt, 1216, col0,      0, kk, lane);
    v16h b1 = load_b_h(Bt, 1216, col0 + 16, 0, kk, lane);
    v16h b2 = load_b_h(Bt, 1216, col0 + 32, 0, kk, lane);
    v16h b3 = load_b_h(Bt, 1216, col0 + 48, 0, kk, lane);
    acc0 = __builtin_amdgcn_wmma_f32_16x16x32_f16(false, a, false, b0, (short)0, acc0, false, false);
    acc1 = __builtin_amdgcn_wmma_f32_16x16x32_f16(false, a, false, b1, (short)0, acc1, false, false);
    acc2 = __builtin_amdgcn_wmma_f32_16x16x32_f16(false, a, false, b2, (short)0, acc2, false, false);
    acc3 = __builtin_amdgcn_wmma_f32_16x16x32_f16(false, a, false, b3, (short)0, acc3, false, false);
  }
#pragma unroll
  for (int i = 0; i < 8; ++i) {
    int m = (lane < 16) ? i : (8 + i);
    float* row = y7 + ((size_t)b*512 + orow + m)*NPTS + (lane & 15);
    row[col0]      = acc0[i];
    row[col0 + 16] = acc1[i];
    row[col0 + 32] = acc2[i];
    row[col0 + 48] = acc3[i];
  }
}

// ---------------- per-channel BN stats from materialized activations ----------------
__global__ __launch_bounds__(256) void chanstats_kernel(const float* __restrict__ y,
                                                        int C, float* __restrict__ st) {
  int o = blockIdx.x;
  int tid = threadIdx.x;
  float sum = 0.f, ss = 0.f;
  for (int s = tid; s < BATCH*NPTS; s += 256) {
    int b = s >> 12, n = s & 4095;
    float v = y[((size_t)b*C + o)*NPTS + n];
    sum += v; ss += v*v;
  }
  __shared__ float S1[256], S2[256];
  S1[tid] = sum; S2[tid] = ss;
  __syncthreads();
  for (int off = 128; off > 0; off >>= 1) {
    if (tid < off) { S1[tid] += S1[tid+off]; S2[tid] += S2[tid+off]; }
    __syncthreads();
  }
  if (tid == 0) {
    float inv = 1.f / (float)(BATCH*NPTS);
    float mean = S1[0] * inv;
    float var  = S2[0] * inv - mean*mean;
    st[2*o] = mean; st[2*o+1] = rsqrtf(var + EPS);
  }
}

// ---------------- w8 WMMA GEMM: z7T (512) -> y8 raw f32 (256) ----------------
__global__ __launch_bounds__(128) void w8_gemm_kernel(const _Float16* __restrict__ z7T,
    const _Float16* __restrict__ w8h, float* __restrict__ y8) {
  int b = blockIdx.x;
  int orow = blockIdx.y * 16;
  int wave = threadIdx.x >> 5, lane = threadIdx.x & 31;
  int col0 = (blockIdx.z * 4 + wave) * 64;
  const _Float16* Bt = z7T + (size_t)b * NPTS * 512;
  const _Float16* bRow = Bt + (size_t)(col0 + (lane & 15)) * 512;
  v8f acc0 = {}, acc1 = {}, acc2 = {}, acc3 = {};
  for (int t = 0; t < 16; ++t) {
    int kk = t * 32;
    __builtin_prefetch((const void*)(bRow + kk + 64), 0, 0);
    v16h a  = load_a_h(w8h, 512, orow, kk, lane);
    v16h b0 = load_b_h(Bt, 512, col0,      0, kk, lane);
    v16h b1 = load_b_h(Bt, 512, col0 + 16, 0, kk, lane);
    v16h b2 = load_b_h(Bt, 512, col0 + 32, 0, kk, lane);
    v16h b3 = load_b_h(Bt, 512, col0 + 48, 0, kk, lane);
    acc0 = __builtin_amdgcn_wmma_f32_16x16x32_f16(false, a, false, b0, (short)0, acc0, false, false);
    acc1 = __builtin_amdgcn_wmma_f32_16x16x32_f16(false, a, false, b1, (short)0, acc1, false, false);
    acc2 = __builtin_amdgcn_wmma_f32_16x16x32_f16(false, a, false, b2, (short)0, acc2, false, false);
    acc3 = __builtin_amdgcn_wmma_f32_16x16x32_f16(false, a, false, b3, (short)0, acc3, false, false);
  }
#pragma unroll
  for (int i = 0; i < 8; ++i) {
    int m = (lane < 16) ? i : (8 + i);
    float* row = y8 + ((size_t)b*256 + orow + m)*NPTS + (lane & 15);
    row[col0]      = acc0[i];
    row[col0 + 16] = acc1[i];
    row[col0 + 32] = acc2[i];
    row[col0 + 48] = acc3[i];
  }
}

// ---------------- w9 WMMA GEMM: z8T (256) -> out (13) ----------------
__global__ __launch_bounds__(128) void w9_gemm_kernel(const _Float16* __restrict__ z8T,
    const _Float16* __restrict__ w9h, float* __restrict__ out) {
  int b = blockIdx.x;
  int wave = threadIdx.x >> 5, lane = threadIdx.x & 31;
  int col0 = (blockIdx.z * 4 + wave) * 64;
  const _Float16* Bt = z8T + (size_t)b * NPTS * 256;
  v8f acc0 = {}, acc1 = {}, acc2 = {}, acc3 = {};
  for (int t = 0; t < 8; ++t) {
    int kk = t * 32;
    v16h a  = load_a_h(w9h, 256, 0, kk, lane);    // rows >=13 pre-zeroed
    v16h b0 = load_b_h(Bt, 256, col0,      0, kk, lane);
    v16h b1 = load_b_h(Bt, 256, col0 + 16, 0, kk, lane);
    v16h b2 = load_b_h(Bt, 256, col0 + 32, 0, kk, lane);
    v16h b3 = load_b_h(Bt, 256, col0 + 48, 0, kk, lane);
    acc0 = __builtin_amdgcn_wmma_f32_16x16x32_f16(false, a, false, b0, (short)0, acc0, false, false);
    acc1 = __builtin_amdgcn_wmma_f32_16x16x32_f16(false, a, false, b1, (short)0, acc1, false, false);
    acc2 = __builtin_amdgcn_wmma_f32_16x16x32_f16(false, a, false, b2, (short)0, acc2, false, false);
    acc3 = __builtin_amdgcn_wmma_f32_16x16x32_f16(false, a, false, b3, (short)0, acc3, false, false);
  }
#pragma unroll
  for (int i = 0; i < 8; ++i) {
    int m = (lane < 16) ? i : (8 + i);
    if (m < 13) {
      float* row = out + ((size_t)b*13 + m)*NPTS + (lane & 15);
      row[col0]      = acc0[i];
      row[col0 + 16] = acc1[i];
      row[col0 + 32] = acc2[i];
      row[col0 + 48] = acc3[i];
    }
  }
}

// ============================ host launch ============================
extern "C" void kernel_launch(void* const* d_in, const int* in_sizes, int n_in,
                              void* d_out, int out_size, void* d_ws, size_t ws_size,
                              hipStream_t stream) {
  (void)in_sizes; (void)n_in; (void)out_size; (void)ws_size;
  const float* x    = (const float*)d_in[0];
  const float* w1   = (const float*)d_in[1];
  const float* w1_1 = (const float*)d_in[2];
  const float* w2   = (const float*)d_in[3];
  const float* w3   = (const float*)d_in[4];
  const float* w3_1 = (const float*)d_in[5];
  const float* w4   = (const float*)d_in[6];
  const float* w5   = (const float*)d_in[7];
  const float* w5_1 = (const float*)d_in[8];
  const float* w5_2 = (const float*)d_in[9];
  const float* w6   = (const float*)d_in[10];
  const float* w7   = (const float*)d_in[11];
  const float* w8   = (const float*)d_in[12];
  const float* w9   = (const float*)d_in[13];

  // workspace carve-up (all offsets 32B-aligned)
  char* ws = (char*)d_ws;
  size_t P = 0;
  int*      idx  = (int*)(ws + P);      P += (size_t)BATCH*NPTS*KMAX*sizeof(int);
  float*    mom  = (float*)(ws + P);    P += (size_t)53760*sizeof(float);  // 3*(342+1056+4160)+37056, padded
  float*    st   = (float*)(ws + P);    P += (size_t)4544*sizeof(float);
  float*    s192 = (float*)(ws + P);    P += (size_t)BATCH*192*NPTS*sizeof(float);
  float*    g    = (float*)(ws + P);    P += (size_t)BATCH*1024*sizeof(float);
  float*    y7   = (float*)(ws + P);    P += (size_t)BATCH*512*NPTS*sizeof(float);
  float*    y8   = (float*)(ws + P);    P += (size_t)BATCH*256*NPTS*sizeof(float);
  _Float16* w6h  = (_Float16*)(ws + P); P += (size_t)1024*192*sizeof(_Float16);
  _Float16* w7h  = (_Float16*)(ws + P); P += (size_t)512*1216*sizeof(_Float16);
  _Float16* w8h  = (_Float16*)(ws + P); P += (size_t)256*512*sizeof(_Float16);
  _Float16* w9h  = (_Float16*)(ws + P); P += (size_t)16*256*sizeof(_Float16);
  _Float16* hT   = (_Float16*)(ws + P); P += (size_t)BATCH*NPTS*1216*sizeof(_Float16);
  _Float16* z7T  = (_Float16*)(ws + P); P += (size_t)BATCH*NPTS*512*sizeof(_Float16);
  _Float16* z8T  = (_Float16*)(ws + P); P += (size_t)BATCH*NPTS*256*sizeof(_Float16);

  const float* Wb[3][3] = {{w1, w1_1, w2}, {w3, w3_1, w4}, {w5, w5_1, w5_2}};
  const int    ks[3]    = {20, 6, 8};
  const int    coff[3]  = {0, 64, 128};

  zero_kernel<<<(53760 + 255)/256, 256, 0, stream>>>(mom, 53760);
  knn_kernel<<<BATCH*NPTS, 32, 0, stream>>>(x, idx);

  // weight conversions to f16
  wconv_kernel<<<(1024*192 + 255)/256, 256, 0, stream>>>(w6, w6h, 1024*192);
  wconv_kernel<<<(512*1216 + 255)/256, 256, 0, stream>>>(w7, w7h, 512*1216);
  wconv_kernel<<<(256*512 + 255)/256, 256, 0, stream>>>(w8, w8h, 256*512);
  w9conv_kernel<<<(16*256 + 255)/256, 256, 0, stream>>>(w9, w9h);

  for (int br = 0; br < 3; ++br) {
    long  S0   = (long)BATCH * NPTS * ks[br];
    float Sinv = 1.f / (float)S0;
    float* m0  = mom + br*5558;          // [324 M][18 mu]
    float* m1  = m0 + 342;               // [1024 M][32 mu]
    float* m2  = m0 + 1398;              // [4096 M][64 mu]
    float* s1  = st + br*320;            // 32*(mean,rstd)
    float* s2  = s1 + 64;                // 64*(mean,rstd)
    float* s3  = s1 + 192;               // 64*(mean,rstd)

    edge_mom0_kernel<<<(int)((S0 + 255)/256), 256, 0, stream>>>(x, idx, ks[br], m0);
    linstats_kernel<<<1, 64, 0, stream>>>(Wb[br][0], 18, 32, m0, Sinv, s1);
    edge_mom1_kernel<<<(int)((S0 + 127)/128), 128, 0, stream>>>(x, idx, ks[br],
                                                                Wb[br][0], s1, m1);
    linstats_kernel<<<1, 64, 0, stream>>>(Wb[br][1], 32, 64, m1, Sinv, s2);
    edge_mom2_kernel<<<(int)((S0 + 127)/128), 128, 0, stream>>>(x, idx, ks[br],
                                                                Wb[br][0], Wb[br][1], s1, s2, m2);
    linstats_kernel<<<1, 64, 0, stream>>>(Wb[br][2], 64, 64, m2, Sinv, s3);
    edge_final_kernel<<<BATCH*NPTS/128, 128, 0, stream>>>(x, idx, ks[br],
        Wb[br][0], Wb[br][1], Wb[br][2], s1, s2, s3, s192, coff[br]);
  }

  float* mom192 = mom + 16674;           // [36864 M][192 mu]
  float* st6 = st + 960;                 // 1024*(mean,rstd)
  float* st7 = st + 3008;                // 512*(mean,rstd)
  float* st8 = st + 4032;                // 256*(mean,rstd)

  s192_to_hT_kernel<<<(BATCH*NPTS*192 + 255)/256, 256, 0, stream>>>(s192, hT);
  mom192_kernel<<<dim3(145, 16), 256, 0, stream>>>(s192, mom192);
  linstats_kernel<<<1024/64, 64, 0, stream>>>(w6, 192, 1024, mom192,
                                              1.f/(float)(BATCH*NPTS), st6);
  w6_max_kernel<<<dim3(BATCH, 64), 128, 0, stream>>>(hT, w6h, st6, g);
  g_to_hT_kernel<<<(BATCH*NPTS*1024 + 255)/256, 256, 0, stream>>>(g, hT);

  w7_gemm_kernel<<<dim3(BATCH, 32, 16), 128, 0, stream>>>(hT, w7h, y7);
  chanstats_kernel<<<512, 256, 0, stream>>>(y7, 512, st7);
  normT_kernel<<<(BATCH*NPTS*512 + 255)/256, 256, 0, stream>>>(y7, st7, 512, z7T);
  w8_gemm_kernel<<<dim3(BATCH, 16, 16), 128, 0, stream>>>(z7T, w8h, y8);
  chanstats_kernel<<<256, 256, 0, stream>>>(y8, 256, st8);
  normT_kernel<<<(BATCH*NPTS*256 + 255)/256, 256, 0, stream>>>(y8, st8, 256, z8T);
  w9_gemm_kernel<<<dim3(BATCH, 1, 16), 128, 0, stream>>>(z8T, w9h, (float*)d_out);
}
